// SNNLIFFunc_86706799772109
// MI455X (gfx1250) — matrix operand order repaired and checked by reading, not compiled, not run on hardware
//
#include <hip/hip_runtime.h>
#include <hip/hip_bf16.h>

// ---------------------------------------------------------------------------
// SNN-LIF fused WMMA implementation for gfx1250 (MI455X).
//
// s@Ws with binary s is EXACT under a bf16 hi/lo split of Ws, so the 31
// recurrent GEMMs run on v_wmma_f32_16x16x32_bf16 with f32 accumulation.
// The one-time x@Wx uses the 3-term bf16 split.
// Staging: Tensor Data Mover (tensor_load_to_lds) with double-buffered LDS,
// hardware row padding (64B row + 16B pad = 80B stride, bank-conflict-free).
// ---------------------------------------------------------------------------

#define Bn 512
#define Dn 2048
#define Tn 32

typedef __attribute__((ext_vector_type(16))) __bf16 v16bf;
typedef __attribute__((ext_vector_type(8)))  __bf16 v8bf;
typedef __attribute__((ext_vector_type(8)))  float  v8f;
typedef __attribute__((ext_vector_type(4)))  unsigned int v4u;
typedef __attribute__((ext_vector_type(8)))  int v8i;
typedef __attribute__((ext_vector_type(4)))  int v4i;

__device__ __forceinline__ v16bf cat8(v8bf a, v8bf b) {
    return __builtin_shufflevector(a, b, 0,1,2,3,4,5,6,7,8,9,10,11,12,13,14,15);
}

// A-fragment (16x32 bf16): lane m holds K {kb..kb+7} and {kb+16..kb+23},
// kb = 0 (lanes 0-15) or 8 (lanes 16-31).  Row pointer is LDS row of m.
__device__ __forceinline__ v16bf load_a_frag(const __bf16* row, int kb) {
    v8bf p0 = *(const v8bf*)(row + kb);
    v8bf p1 = *(const v8bf*)(row + kb + 16);
    return cat8(p0, p1);
}

// B-fragment (32x16 bf16): lane n holds K {kb..kb+7} and {kb+16..kb+23}, kb = 0 or 8.
__device__ __forceinline__ v16bf load_b_frag(const __bf16* row, int kb) {
    v8bf p0 = *(const v8bf*)(row + kb);
    v8bf p1 = *(const v8bf*)(row + kb + 16);
    return cat8(p0, p1);
}

// TDM 2D tile load: tile_rows x 32 bf16 tile from a row-major [*, 2048] bf16
// tensor into LDS, with 16B padding appended to every 64B row (80B stride).
// D# per CDNA5 ISA ch.8: group0 = {flags, lds_addr, global_addr, type},
// group1 = {pad cfg + dims/strides}.  Groups 2/3 zero (2D tensor).
// This toolchain exposes the 6-arg builtin (extra int32x8 group + cpol).
__device__ __forceinline__ void tdm_load_2d(unsigned lds_off,
                                            const __bf16* gptr,
                                            int tile_rows) {
    unsigned long long ga = (unsigned long long)gptr;
    v4u g0;
    g0.x = 1u;                                   // count=1, user descriptor
    g0.y = lds_off;                              // LDS byte address
    g0.z = (unsigned)ga;                         // global_addr[31:0]
    g0.w = (unsigned)((ga >> 32) & 0x01FFFFFFu)  // global_addr[56:32]
           | 0x80000000u;                        // type=2 ("image")
    v8i g1;
    g1[0] = 0x06D10000;        // data_size=2B | pad_enable | pad every 16dw, 4dw
    g1[1] = (int)(2048u << 16);// tensor_dim0=2048 (low16 -> bits 63:48)
    g1[2] = (int)(2048u << 16);// tensor_dim0 hi=0 | tensor_dim1=2048 low16
    g1[3] = (int)(32u << 16);  // tensor_dim1 hi=0 | tile_dim0=32
    g1[4] = tile_rows;         // tile_dim1 | tile_dim2=0
    g1[5] = 2048;              // tensor_dim0_stride (elements)
    g1[6] = 0;
    g1[7] = 0;
    v4i z4 = {0, 0, 0, 0};
    v8i z8 = {0, 0, 0, 0, 0, 0, 0, 0};
    __builtin_amdgcn_tensor_load_to_lds(g0, g1, z4, z4, z8, 0);
}

// --------------------------- prep kernels ----------------------------------

// W (row-major [k][n]) -> WT_hi/WT_lo (bf16, column-major: WT[n*D + k]).
__global__ __launch_bounds__(256) void split_transpose_kernel(
        const float* __restrict__ W,
        __bf16* __restrict__ WT_hi, __bf16* __restrict__ WT_lo) {
    size_t o = (size_t)blockIdx.x * 256 + threadIdx.x;   // o = n*D + k
    int n = (int)(o / Dn);
    int k = (int)(o % Dn);
    float w = W[(size_t)k * Dn + n];
    __bf16 h = (__bf16)w;
    WT_hi[o] = h;
    WT_lo[o] = (__bf16)(w - (float)h);
}

// x -> x_hi, x_lo (row-major, bf16)
__global__ __launch_bounds__(256) void split_kernel(
        const float* __restrict__ X,
        __bf16* __restrict__ X_hi, __bf16* __restrict__ X_lo) {
    size_t o = (size_t)blockIdx.x * 256 + threadIdx.x;
    float w = X[o];
    __bf16 h = (__bf16)w;
    X_hi[o] = h;
    X_lo[o] = (__bf16)(w - (float)h);
}

// --------------------------- fused GEMM + LIF ------------------------------
//
// Block tile: M=128, N=64.  8 waves: wave (wm,wn) with wm in 0..3 (32 rows),
// wn in 0..1 (32 cols); each wave holds a 2x2 grid of 16x16 f32 accumulators.
// K is consumed in 32-wide chunks, double-buffered via TDM.
//
// mode 0 (init):  acc = A0@Bhi + A0@Blo + A1@Bhi   (A0=x_hi, A1=x_lo)
// mode 1 (step):  acc = A0@Bhi + A0@Blo            (A0=s_prev, exact)
__global__ __launch_bounds__(256) void lif_gemm_kernel(
        const __bf16* __restrict__ A0,
        const __bf16* __restrict__ A1,
        const __bf16* __restrict__ BT_hi,
        const __bf16* __restrict__ BT_lo,
        const float*  __restrict__ bias,
        const float*  __restrict__ x1_in,
        float*        __restrict__ x1_out,
        float*        __restrict__ u,
        float*        __restrict__ a,
        __bf16*       __restrict__ s_out,
        float*        __restrict__ out,
        int mode, int last, float inv_denom) {

    // [buf][mat][row][40 halves]: 64B data + 16B pad per row (TDM-padded)
    __shared__ __align__(16) __bf16 lA[2][2][128][40];   // 40 KB
    __shared__ __align__(16) __bf16 lB[2][2][64][40];    // 20 KB

    const int tid  = threadIdx.x;
    const int lane = tid & 31;
    const int wid  = tid >> 5;
    const int wm   = wid & 3;            // 0..3 -> 32-row slice
    const int wn   = wid >> 2;           // 0..1 -> 32-col slice
    const int m0   = blockIdx.y * 128;
    const int n0   = blockIdx.x * 64;

    const int lm   = lane & 15;
    const int kb_a = (lane >> 4) << 3;   // 0 / 8
    const int kb_b = (lane >> 4) << 3;   // 0 / 8

    v8f acc[2][2] = {};

    const int NCHUNK = Dn / 32;

    // issue chunk 0 into buffer 0 (one wave drives the TDM; EXEC-independent)
    if (wid == 0) {
        tdm_load_2d((unsigned)(unsigned long long)&lA[0][0][0][0],
                    A0 + (size_t)m0 * Dn, 128);
        if (mode == 0)
            tdm_load_2d((unsigned)(unsigned long long)&lA[0][1][0][0],
                        A1 + (size_t)m0 * Dn, 128);
        tdm_load_2d((unsigned)(unsigned long long)&lB[0][0][0][0],
                    BT_hi + (size_t)n0 * Dn, 64);
        tdm_load_2d((unsigned)(unsigned long long)&lB[0][1][0][0],
                    BT_lo + (size_t)n0 * Dn, 64);
    }

    for (int c = 0; c < NCHUNK; ++c) {
        const int buf = c & 1;
        if (wid == 0) {
            if (c + 1 < NCHUNK) {
                // prefetch chunk c+1 into the other buffer, then wait for
                // chunk c only (TDM ops complete in issue order per wave)
                const int nb = buf ^ 1;
                const int k1 = (c + 1) * 32;
                tdm_load_2d((unsigned)(unsigned long long)&lA[nb][0][0][0],
                            A0 + (size_t)m0 * Dn + k1, 128);
                if (mode == 0) {
                    tdm_load_2d((unsigned)(unsigned long long)&lA[nb][1][0][0],
                                A1 + (size_t)m0 * Dn + k1, 128);
                    tdm_load_2d((unsigned)(unsigned long long)&lB[nb][0][0][0],
                                BT_hi + (size_t)n0 * Dn + k1, 64);
                    tdm_load_2d((unsigned)(unsigned long long)&lB[nb][1][0][0],
                                BT_lo + (size_t)n0 * Dn + k1, 64);
                    __builtin_amdgcn_s_wait_tensorcnt((short)4);
                } else {
                    tdm_load_2d((unsigned)(unsigned long long)&lB[nb][0][0][0],
                                BT_hi + (size_t)n0 * Dn + k1, 64);
                    tdm_load_2d((unsigned)(unsigned long long)&lB[nb][1][0][0],
                                BT_lo + (size_t)n0 * Dn + k1, 64);
                    __builtin_amdgcn_s_wait_tensorcnt((short)3);
                }
            } else {
                __builtin_amdgcn_s_wait_tensorcnt((short)0);
            }
        }
        __syncthreads();   // chunk c visible to all waves

        // ---- fragments ----
        v16bf afr0[2], afr1[2], bfr[2][2];
        #pragma unroll
        for (int sm = 0; sm < 2; ++sm) {
            int m = wm * 32 + sm * 16 + lm;
            afr0[sm] = load_a_frag(&lA[buf][0][m][0], kb_a);
        }
        if (mode == 0) {
            #pragma unroll
            for (int sm = 0; sm < 2; ++sm) {
                int m = wm * 32 + sm * 16 + lm;
                afr1[sm] = load_a_frag(&lA[buf][1][m][0], kb_a);
            }
        }
        #pragma unroll
        for (int h = 0; h < 2; ++h)
            #pragma unroll
            for (int sn = 0; sn < 2; ++sn) {
                int n = wn * 32 + sn * 16 + lm;
                bfr[h][sn] = load_b_frag(&lB[buf][h][n][0], kb_b);
            }

        // ---- matrix cores ----
        #pragma unroll
        for (int sm = 0; sm < 2; ++sm)
            #pragma unroll
            for (int sn = 0; sn < 2; ++sn) {
                acc[sm][sn] = __builtin_amdgcn_wmma_f32_16x16x32_bf16(
                    false, afr0[sm], false, bfr[0][sn], (short)0,
                    acc[sm][sn], false, false);
                acc[sm][sn] = __builtin_amdgcn_wmma_f32_16x16x32_bf16(
                    false, afr0[sm], false, bfr[1][sn], (short)0,
                    acc[sm][sn], false, false);
            }
        if (mode == 0) {
            #pragma unroll
            for (int sm = 0; sm < 2; ++sm)
                #pragma unroll
                for (int sn = 0; sn < 2; ++sn)
                    acc[sm][sn] = __builtin_amdgcn_wmma_f32_16x16x32_bf16(
                        false, afr1[sm], false, bfr[0][sn], (short)0,
                        acc[sm][sn], false, false);
        }
        __syncthreads();   // buf free for chunk c+2's TDM
    }

    // ---- LIF epilogue (C/D layout: VGPR i -> M = i (+8 for lanes 16-31)) --
    const float L = 0.9f;
    const int gm_base = m0 + wm * 32;
    const int gn_base = n0 + wn * 32;
    const int moff = (lane >> 4) << 3;
    const int ncol = lane & 15;

    #pragma unroll
    for (int sm = 0; sm < 2; ++sm)
        #pragma unroll
        for (int sn = 0; sn < 2; ++sn)
            #pragma unroll
            for (int i = 0; i < 8; ++i) {
                int m = gm_base + sm * 16 + i + moff;
                int n = gn_base + sn * 16 + ncol;
                size_t idx = (size_t)m * Dn + n;
                float accv = acc[sm][sn][i];
                if (mode == 0) {
                    float x1v = accv + bias[n];
                    float sv  = (x1v >= 1.0f) ? 1.0f : 0.0f;
                    x1_out[idx] = x1v;
                    u[idx] = (x1v - sv) * L;
                    a[idx] = sv;
                    s_out[idx] = (__bf16)sv;
                } else {
                    float uv = u[idx] + accv + bias[n] + x1_in[idx];
                    float sv = (uv >= 1.0f) ? 1.0f : 0.0f;
                    uv = (uv - sv) * L;
                    float av = a[idx] * L + sv;
                    u[idx] = uv;
                    a[idx] = av;
                    s_out[idx] = (__bf16)sv;
                    if (last) out[idx] = av * inv_denom;
                }
            }
}

// --------------------------- host launcher ---------------------------------

extern "C" void kernel_launch(void* const* d_in, const int* in_sizes, int n_in,
                              void* d_out, int out_size, void* d_ws, size_t ws_size,
                              hipStream_t stream) {
    const float* x  = (const float*)d_in[0];
    const float* Wx = (const float*)d_in[1];
    const float* bx = (const float*)d_in[2];
    const float* Ws = (const float*)d_in[3];
    const float* bs = (const float*)d_in[4];
    // d_in[5] = time_step (device scalar); fixed at Tn=32 per problem spec.

    char* ws = (char*)d_ws;
    const size_t WB = (size_t)Dn * Dn * 2;   // bf16 weight matrix bytes
    const size_t XB = (size_t)Bn * Dn * 2;   // bf16 state bytes
    const size_t FB = (size_t)Bn * Dn * 4;   // f32 state bytes

    __bf16* WxT_hi = (__bf16*)(ws);
    __bf16* WxT_lo = (__bf16*)(ws + WB);
    __bf16* WsT_hi = (__bf16*)(ws + 2 * WB);
    __bf16* WsT_lo = (__bf16*)(ws + 3 * WB);
    __bf16* x_hi   = (__bf16*)(ws + 4 * WB);
    __bf16* x_lo   = (__bf16*)(ws + 4 * WB + XB);
    float*  x1     = (float*) (ws + 4 * WB + 2 * XB);
    float*  u      = (float*) (ws + 4 * WB + 2 * XB + FB);
    float*  a      = (float*) (ws + 4 * WB + 2 * XB + 2 * FB);
    __bf16* sbuf0  = (__bf16*)(ws + 4 * WB + 2 * XB + 3 * FB);
    __bf16* sbuf1  = (__bf16*)(ws + 4 * WB + 2 * XB + 3 * FB + XB);

    float* out = (float*)d_out;

    float lp = 1.0f;
    for (int i = 0; i < Tn; ++i) lp *= 0.9f;
    const float inv_denom = (1.0f - 0.9f) / (1.0f - lp);

    // one-time bf16 hi/lo splits (weights transposed to column-major)
    split_transpose_kernel<<<(Dn * (size_t)Dn) / 256, 256, 0, stream>>>(Wx, WxT_hi, WxT_lo);
    split_transpose_kernel<<<(Dn * (size_t)Dn) / 256, 256, 0, stream>>>(Ws, WsT_hi, WsT_lo);
    split_kernel<<<((size_t)Bn * Dn) / 256, 256, 0, stream>>>(x, x_hi, x_lo);

    dim3 gg(Dn / 64, Bn / 128);   // 32 x 4 = 128 workgroups

    // init: x1 = x@Wx + bx (3-term bf16 split), LIF init, spikes -> sbuf0
    lif_gemm_kernel<<<gg, 256, 0, stream>>>(
        x_hi, x_lo, WxT_hi, WxT_lo, bx,
        nullptr, x1, u, a, sbuf0, out, /*mode=*/0, /*last=*/0, inv_denom);

    // 31 recurrent steps: s@Ws exact via hi/lo split; spike double-buffered
    for (int t = 1; t < Tn; ++t) {
        const __bf16* sp = (t & 1) ? sbuf0 : sbuf1;
        __bf16*       sn = (t & 1) ? sbuf1 : sbuf0;
        lif_gemm_kernel<<<gg, 256, 0, stream>>>(
            sp, nullptr, WsT_hi, WsT_lo, bs,
            x1, nullptr, u, a, sn, out, /*mode=*/1,
            /*last=*/(t == Tn - 1) ? 1 : 0, inv_denom);
    }
}
